// GCN_33036888441456
// MI455X (gfx1250) — matrix operand order repaired and checked by reading, not compiled, not run on hardware
//
#include <hip/hip_runtime.h>
#include <hip/hip_bf16.h>

typedef __attribute__((ext_vector_type(16))) __bf16 v16bf;
typedef __attribute__((ext_vector_type(8)))  float  v8f;

#define F_IN 64
#define HID  3
#define F_OUT 256
#define F_TR 512

// ---------------- utility ----------------
__global__ void k_zero(float* p, int n) {
    int i = blockIdx.x * 256 + threadIdx.x;
    if (i < n) p[i] = 0.f;
}

// ---------------- degree / norm ----------------
__global__ void k_degree(const int* ei, int E, int Nn, float* deg) {
    int e = blockIdx.x * 256 + threadIdx.x;
    if (e >= E) return;
    int s = ei[e], d = ei[E + e];
    if (s == d) return;
    if ((unsigned)s >= (unsigned)Nn || (unsigned)d >= (unsigned)Nn) return;
    atomicAdd(&deg[s], 1.0f);
}

__global__ void k_dinv(const float* deg, float* dinv, int Nn) {
    int i = blockIdx.x * 256 + threadIdx.x;
    if (i >= Nn) return;
    float dg = deg[i];
    dinv[i] = (dg > 0.f) ? rsqrtf(fmaxf(dg, 1.0f)) : 0.f;
}

// ---------------- GEMM1: xa = x @ [W0a | W1a], [N,64] x [64,6] ----------------
__global__ void k_gemm1(const float* __restrict__ x,
                        const float* __restrict__ W0a, const float* __restrict__ W1a,
                        float* __restrict__ xa0, float* __restrict__ xa1, int Nn) {
    __shared__ float xs[128 * 65];   // padded stride: bank-conflict-free
    __shared__ float Ws[64 * 6];
    int tid = threadIdx.x;
    int n0 = blockIdx.x * 128;
    for (int i = tid; i < 64 * 6; i += 256) {
        int r = i / 6, j = i % 6;
        Ws[i] = (j < 3) ? W0a[r * 3 + j] : W1a[r * 3 + (j - 3)];
    }
    for (int i = tid; i < 128 * 64; i += 256) {   // coalesced global reads
        int m = i >> 6, f = i & 63;
        int node = n0 + m;
        xs[m * 65 + f] = (node < Nn) ? x[node * 64 + f] : 0.f;
    }
    __syncthreads();
    if (tid < 128) {
        int node = n0 + tid;
        if (node < Nn) {
            float a0 = 0, a1 = 0, a2 = 0, a3 = 0, a4 = 0, a5 = 0;
#pragma unroll 16
            for (int i = 0; i < 64; ++i) {
                float v = xs[tid * 65 + i];
                a0 = fmaf(v, Ws[i * 6 + 0], a0);
                a1 = fmaf(v, Ws[i * 6 + 1], a1);
                a2 = fmaf(v, Ws[i * 6 + 2], a2);
                a3 = fmaf(v, Ws[i * 6 + 3], a3);
                a4 = fmaf(v, Ws[i * 6 + 4], a4);
                a5 = fmaf(v, Ws[i * 6 + 5], a5);
            }
            xa0[node * 3 + 0] = a0; xa0[node * 3 + 1] = a1; xa0[node * 3 + 2] = a2;
            xa1[node * 3 + 0] = a3; xa1[node * 3 + 1] = a4; xa1[node * 3 + 2] = a5;
        }
    }
}

// ---------------- edge scatter: acc[dst] += norm(e) * feat[src] (width 3) ----------------
__global__ void k_scatter(const int* __restrict__ ei, int E, int Nn,
                          const float* __restrict__ dinv,
                          const float* __restrict__ feat, float* __restrict__ acc) {
    int e = blockIdx.x * 256 + threadIdx.x;
    if (e >= E) return;
    int s = ei[e], d = ei[E + e];
    if (s == d) return;
    if ((unsigned)s >= (unsigned)Nn || (unsigned)d >= (unsigned)Nn) return;
    float nm = -dinv[s] * dinv[d];
    if (nm == 0.f) return;
    atomicAdd(&acc[d * 3 + 0], nm * feat[s * 3 + 0]);
    atomicAdd(&acc[d * 3 + 1], nm * feat[s * 3 + 1]);
    atomicAdd(&acc[d * 3 + 2], nm * feat[s * 3 + 2]);
}

// ---------------- h1 = relu(xa0 + t1 + b1) ----------------
__global__ void k_relu1(const float* xa0, const float* t1, const float* b1,
                        float* h1, int Nn) {
    int i = blockIdx.x * 256 + threadIdx.x;
    if (i >= 3 * Nn) return;
    int j = i - (i / 3) * 3;
    h1[i] = fmaxf(xa0[i] + t1[i] + b1[j], 0.f);
}

// ---------------- pre-pack Wl (fp32 [256][512]) into bf16 WMMA B-fragment order ----------
// B (32x16 per frag): lanes 0-15 hold K=0..7,16..23 (col = lane&15), lanes 16-31 hold K=8..15,24..31.
__global__ void k_packWl(const float* __restrict__ Wl, __bf16* __restrict__ Wlp) {
    int idx = blockIdx.x * 256 + threadIdx.x;
    if (idx >= F_OUT * F_TR) return;
    int k = idx >> 9, n = idx & 511;
    float v = Wl[idx];
    int kt = k >> 5, ko = k & 31, nt = n >> 4, nn = n & 15;
    int lane = nn + ((ko & 8) << 1);
    int elem = (ko & 7) | ((ko >> 4) << 3);
    Wlp[((((nt << 3) + kt) << 5) + lane) * 16 + elem] = (__bf16)v;
}

// ---------------- fused GEMM2 + ReLU + GEMM3 + bias (WMMA) ----------------
// Block = 256 threads (8 wave32), owns 64 nodes. h2 (64x256) built in LDS in
// A-fragment layout (bf16); 8 waves do 128 16x16 output tiles via
// v_wmma_f32_16x16x32_bf16 (K=256 -> 8 wmma per tile), fp32 accumulate.
// Phase 3 runs two independent accumulator chains (two N-tiles) per iteration
// so B-fragment loads of one chain overlap WMMAs of the other.
__global__ void __launch_bounds__(256)
k_head(const float* __restrict__ h1, const float* __restrict__ t2,
       const float* __restrict__ W0b, const float* __restrict__ W1b,
       const float* __restrict__ b2,
       const __bf16* __restrict__ Wlp, const float* __restrict__ bl,
       float* __restrict__ out, int Nn) {
    __shared__ float h1s[64][3];
    __shared__ float t2s[64][3];
    __shared__ __align__(32) __bf16 afrag[4][8][32][16];  // [mtile][ktile][lane][elem] = 32 KB

    int tid = threadIdx.x;
    int n0 = blockIdx.x * 64;

    // Phase 1: stage the 64x6 input tile
    for (int i = tid; i < 64 * 3 * 2; i += 256) {
        int half = i / 192, r = i % 192;
        int m = r / 3, j = r % 3;
        int node = n0 + m;
        float v = 0.f;
        if (node < Nn) v = half ? t2[node * 3 + j] : h1[node * 3 + j];
        if (half) t2s[m][j] = v; else h1s[m][j] = v;
    }
    __syncthreads();

    // Phase 2: h2 = relu([h1|t2] @ [[W0b],[W1b]] + b2) -> bf16 A fragments in LDS
    {
        int c = tid;                              // output column 0..255
        float w00 = W0b[c], w01 = W0b[256 + c], w02 = W0b[512 + c];
        float w10 = W1b[c], w11 = W1b[256 + c], w12 = W1b[512 + c];
        float bb = b2[c];
        int kt = c >> 5, ko = c & 31;
        int laneAdd = (ko & 8) << 1;              // +16 lanes for K 8-15 / 24-31
        int elem = (ko & 7) | ((ko >> 4) << 3);   // ISA 16-bit A-matrix layout
        for (int m = 0; m < 64; ++m) {
            float v = fmaf(h1s[m][0], w00, fmaf(h1s[m][1], w01, fmaf(h1s[m][2], w02,
                      fmaf(t2s[m][0], w10, fmaf(t2s[m][1], w11, fmaf(t2s[m][2], w12, bb))))));
            v = v > 0.f ? v : 0.f;
            afrag[m >> 4][kt][(m & 15) + laneAdd][elem] = (__bf16)v;
        }
    }
    __syncthreads();

    // Phase 3: out[64 x 512] = h2 @ Wl + bl via WMMA
    int wave = tid >> 5, lane = tid & 31;
    int mt = wave >> 1;                // 2 waves per M-tile
    int ntb = (wave & 1) << 4;         // each wave: 16 N-tiles (processed in pairs)
    v16bf A[8];
#pragma unroll
    for (int kt = 0; kt < 8; ++kt)
        A[kt] = *(const v16bf*)&afrag[mt][kt][lane][0];

    const __bf16* Bl = Wlp + lane * 16;          // per-lane fragment base
    int rowbase = n0 + (mt << 4) + ((lane >> 4) << 3);  // D layout: lanes>=16 -> M+8
    int colh = lane & 15;
    const bool full = (n0 + 64 <= Nn);           // uniform: 1562 of 1563 blocks

    for (int t = 0; t < 8; ++t) {
        int nt0 = ntb + 2 * t, nt1 = nt0 + 1;
        // warm WGP cache for next iteration's fragments (global_prefetch_b8)
        __builtin_prefetch(Bl + (size_t)(nt0 + 2) * 8 * 512, 0, 1);
        __builtin_prefetch(Bl + (size_t)(nt1 + 2) * 8 * 512, 0, 1);

        float bias0 = bl[(nt0 << 4) + colh];
        float bias1 = bl[(nt1 << 4) + colh];
        v8f acc0, acc1;
#pragma unroll
        for (int i = 0; i < 8; ++i) { acc0[i] = bias0; acc1[i] = bias1; }
#pragma unroll
        for (int kt = 0; kt < 8; ++kt) {
            v16bf B0 = *(const v16bf*)(Bl + (size_t)((nt0 << 3) + kt) * 512);
            v16bf B1 = *(const v16bf*)(Bl + (size_t)((nt1 << 3) + kt) * 512);
            acc0 = __builtin_amdgcn_wmma_f32_16x16x32_bf16(
                       false, A[kt], false, B0, (short)0, acc0, false, false);
            acc1 = __builtin_amdgcn_wmma_f32_16x16x32_bf16(
                       false, A[kt], false, B1, (short)0, acc1, false, false);
        }
        float* o0 = out + (size_t)rowbase * 512 + (nt0 << 4) + colh;
        float* o1 = out + (size_t)rowbase * 512 + (nt1 << 4) + colh;
        if (full) {
            // streaming output: non-temporal, never re-read
#pragma unroll
            for (int v = 0; v < 8; ++v) {
                __builtin_nontemporal_store(acc0[v], o0 + (size_t)v * 512);
                __builtin_nontemporal_store(acc1[v], o1 + (size_t)v * 512);
            }
        } else {
#pragma unroll
            for (int v = 0; v < 8; ++v) {
                if (rowbase + v < Nn) {
                    __builtin_nontemporal_store(acc0[v], o0 + (size_t)v * 512);
                    __builtin_nontemporal_store(acc1[v], o1 + (size_t)v * 512);
                }
            }
        }
    }
}

// ---------------- launch ----------------
extern "C" void kernel_launch(void* const* d_in, const int* in_sizes, int n_in,
                              void* d_out, int out_size, void* d_ws, size_t ws_size,
                              hipStream_t stream) {
    const float* x   = (const float*)d_in[0];
    const int*   ei  = (const int*)d_in[1];
    const float* W0a = (const float*)d_in[2];
    const float* W1a = (const float*)d_in[3];
    const float* b1  = (const float*)d_in[4];
    const float* W0b = (const float*)d_in[5];
    const float* W1b = (const float*)d_in[6];
    const float* b2  = (const float*)d_in[7];
    const float* Wl  = (const float*)d_in[8];
    const float* bl  = (const float*)d_in[9];
    float* out = (float*)d_out;

    const int Nn = in_sizes[0] / F_IN;
    const int E  = in_sizes[1] / 2;

    // workspace layout (floats)
    float* ws   = (float*)d_ws;
    float* deg  = ws;                  // N
    float* dinv = ws + (size_t)Nn;     // N
    float* xa0  = ws + (size_t)2 * Nn; // 3N
    float* xa1  = ws + (size_t)5 * Nn; // 3N
    float* h1   = ws + (size_t)8 * Nn; // 3N
    float* t1   = ws + (size_t)11 * Nn;// 3N
    float* t2   = ws + (size_t)14 * Nn;// 3N  (t1,t2 contiguous for one zero pass)
    size_t wlpOff = (((size_t)17 * Nn * sizeof(float)) + 255) & ~(size_t)255;
    __bf16* Wlp = (__bf16*)((char*)d_ws + wlpOff);  // 256 KB, 256B aligned

    const int TB = 256;
    // zero accumulators
    k_zero<<<(Nn + TB - 1) / TB, TB, 0, stream>>>(deg, Nn);
    k_zero<<<(6 * Nn + TB - 1) / TB, TB, 0, stream>>>(t1, 6 * Nn);
    // pack Wl -> bf16 B fragments (independent)
    k_packWl<<<(F_OUT * F_TR + TB - 1) / TB, TB, 0, stream>>>(Wl, Wlp);
    // graph normalization
    k_degree<<<(E + TB - 1) / TB, TB, 0, stream>>>(ei, E, Nn, deg);
    k_dinv<<<(Nn + TB - 1) / TB, TB, 0, stream>>>(deg, dinv, Nn);
    // layer 1: project first, then scatter in width-3 space
    k_gemm1<<<(Nn + 127) / 128, TB, 0, stream>>>(x, W0a, W1a, xa0, xa1, Nn);
    k_scatter<<<(E + TB - 1) / TB, TB, 0, stream>>>(ei, E, Nn, dinv, xa1, t1);
    k_relu1<<<(3 * Nn + TB - 1) / TB, TB, 0, stream>>>(xa0, t1, b1, h1, Nn);
    // layer 2 propagation (width 3), head GEMMs fused with WMMA
    k_scatter<<<(E + TB - 1) / TB, TB, 0, stream>>>(ei, E, Nn, dinv, h1, t2);
    k_head<<<(Nn + 63) / 64, TB, 0, stream>>>(h1, t2, W0b, W1b, b2, Wlp, bl, out, Nn);
}
